// ATAE_LSTM_47450798686243
// MI455X (gfx1250) — compile-verified, matrix operand order checked by reading, not run on hardware
//
#include <hip/hip_runtime.h>
#include <hip/hip_bf16.h>
#include <stdint.h>

// ---------------------------------------------------------------------------
// ATAE-LSTM forward for MI455X (gfx1250), wave32 + v_wmma_f32_16x16x32_bf16.
// All heavy GEMMs run on the WMMA pipe with bf16 inputs / f32 accumulation.
// ---------------------------------------------------------------------------

#define E_   512
#define HD_  1024
#define B_   128
#define L_   256
#define D_   1024       // 2*E
#define HA_  1536       // HD + E
#define G4_  4096       // 4*HD
#define BL_  (B_ * L_)  // 32768

typedef __attribute__((ext_vector_type(16))) __bf16 v16bf;
typedef __attribute__((ext_vector_type(8)))  float  v8f;

__device__ __forceinline__ uint16_t f2bf(float f) {
  union { float f; uint32_t u; } v; v.f = f;
  uint32_t u = v.u;
  u += 0x7FFFu + ((u >> 16) & 1u);   // round-to-nearest-even
  return (uint16_t)(u >> 16);
}
__device__ __forceinline__ float bf2f(uint16_t h) {
  union { uint32_t u; float f; } v; v.u = ((uint32_t)h) << 16;
  return v.f;
}
__device__ __forceinline__ float sigmoidf(float x) {
  return 1.0f / (1.0f + __expf(-x));
}

// Load one 16x32 bf16 WMMA fragment. Per CDNA5 ISA layout (CDNA5 ISA),
// lane l holds row (passed in), K-chunks [k0+c, k0+c+8) and [k0+c+16, k0+c+24)
// with c = 8*(l>=16). Two 16B loads -> global_load_b128 pair.
__device__ __forceinline__ v16bf load_frag(const uint16_t* __restrict__ base,
                                           long ld, int row, int k0) {
  int c = ((threadIdx.x >> 4) & 1) << 3;
  const uint16_t* p = base + (long)row * ld + k0 + c;
  union { v16bf bf; uint4 u[2]; } f;
  f.u[0] = *(const uint4*)(p);
  f.u[1] = *(const uint4*)(p + 16);
  return f.bf;
}

template <int MT>
struct FragBuf { v16bf a[MT]; v16bf b[4]; };

template <int MT>
__device__ __forceinline__ void load_step(const uint16_t* __restrict__ A, long ldA,
                                          int rshift,
                                          const uint16_t* __restrict__ Bm, long ldB,
                                          int kb0, int m0, int n0, int lr, int k,
                                          FragBuf<MT>& f) {
#pragma unroll
  for (int i = 0; i < MT; ++i)
    f.a[i] = load_frag(A, ldA, (m0 + 16 * i + lr) >> rshift, k);
#pragma unroll
  for (int j = 0; j < 4; ++j)
    f.b[j] = load_frag(Bm, ldB, n0 + (j << 4) + lr, kb0 + k);
}

template <int MT>
__device__ __forceinline__ void mma_step(const FragBuf<MT>& f, v8f (&acc)[MT][4]) {
#pragma unroll
  for (int j = 0; j < 4; ++j)
#pragma unroll
    for (int i = 0; i < MT; ++i)
      acc[i][j] = __builtin_amdgcn_wmma_f32_16x16x32_bf16(
          false, f.a[i], false, f.b[j], (short)0, acc[i][j], false, false);
}

// One K-segment of a (16*MT)x64 tile: ping-pong double-buffered and unrolled
// x2, so fragment loads for step k+32 issue while WMMAs for step k run, with
// no inter-buffer register copies. Requires Kseg % 64 == 0 (or Kseg == 0),
// which holds for every GEMM in this model (1024 / 512 / 1536 / 256).
template <int MT>
__device__ __forceinline__ void gemm_seg(const uint16_t* __restrict__ A, long ldA,
                                         int rshift,
                                         const uint16_t* __restrict__ Bm, long ldB,
                                         int kb0, int Kseg,
                                         int m0, int n0, int lr,
                                         v8f (&acc)[MT][4]) {
  if (Kseg <= 0) return;
  FragBuf<MT> f0, f1;
  load_step<MT>(A, ldA, rshift, Bm, ldB, kb0, m0, n0, lr, 0, f0);
  int s = 32;
  for (; s + 32 < Kseg; s += 64) {
    load_step<MT>(A, ldA, rshift, Bm, ldB, kb0, m0, n0, lr, s, f1);
    mma_step<MT>(f0, acc);
    load_step<MT>(A, ldA, rshift, Bm, ldB, kb0, m0, n0, lr, s + 32, f0);
    mma_step<MT>(f1, acc);
  }
  load_step<MT>(A, ldA, rshift, Bm, ldB, kb0, m0, n0, lr, s, f1);  // s == Kseg-32
  mma_step<MT>(f0, acc);
  mma_step<MT>(f1, acc);
}

// Generic WMMA GEMM:  C(M x N) = concatK(A1, A2) @ B^T + bias, opt. tanh.
//   A rows: index m; for k <  Ksplit read A1[m][k]            (stride ld1)
//           for k >= Ksplit read A2[m >> rshift][k - Ksplit]  (stride ld2)
//   B is row-major (N x Ktot), i.e. the weight matrix as stored.
template <int MT>
__global__ __launch_bounds__(32)
void gemm_wmma(const uint16_t* __restrict__ A1, long ld1,
               const uint16_t* __restrict__ A2, long ld2, int rshift, int Ksplit,
               const uint16_t* __restrict__ Bm, long ldB,
               const float* __restrict__ bias, int Ktot,
               float* __restrict__ Cf, uint16_t* __restrict__ Cb, long ldC,
               int act /*0=none, 1=tanh*/) {
  const int m0 = blockIdx.x * (16 * MT);
  const int n0 = blockIdx.y << 6;
  const int l  = threadIdx.x & 31;
  const int lr = l & 15;

  v8f acc[MT][4] = {};
  gemm_seg<MT>(A1, ld1, 0,      Bm, ldB, 0,      Ksplit,        m0, n0, lr, acc);
  gemm_seg<MT>(A2, ld2, rshift, Bm, ldB, Ksplit, Ktot - Ksplit, m0, n0, lr, acc);

  // C/D layout: element r of v8f = C[m0 + 16i + r + 8*(l>=16)][n0 + 16j + lr]
#pragma unroll
  for (int i = 0; i < MT; ++i) {
    const int rbase = m0 + 16 * i + (((l >> 4) & 1) << 3);
#pragma unroll
    for (int j = 0; j < 4; ++j) {
      const int col = n0 + (j << 4) + lr;
      const float bv = bias ? bias[col] : 0.0f;
#pragma unroll
      for (int r = 0; r < 8; ++r) {
        float v = acc[i][j][r] + bv;
        if (act) v = tanhf(v);
        long off = (long)(rbase + r) * ldC + col;
        if (Cf) Cf[off] = v;
        if (Cb) Cb[off] = f2bf(v);
      }
    }
  }
}

// LSTM pointwise gates: g(128x4096) -> h, c; writes Hs (f32, output) and
// bf16 copies of h (recurrent input + attention input).
__global__ void lstm_gates(const float* __restrict__ g, float* __restrict__ c,
                           float* __restrict__ HsF, uint16_t* __restrict__ hbf,
                           uint16_t* __restrict__ HsB, int t) {
  int idx = blockIdx.x * blockDim.x + threadIdx.x;
  if (idx >= B_ * HD_) return;
  int b = idx >> 10, i = idx & 1023;
  const float* gr = g + (long)b * G4_;
  float gi = gr[i], gf = gr[i + 1024], gg = gr[i + 2048], go = gr[i + 3072];
  float cn = sigmoidf(gf) * c[idx] + sigmoidf(gi) * tanhf(gg);
  float h  = sigmoidf(go) * tanhf(cn);
  c[idx] = cn;
  long ho = (long)b * (L_ * HD_) + (long)t * HD_ + i;
  HsF[ho] = h;
  HsB[ho] = f2bf(h);
  hbf[idx] = f2bf(h);
}

// Row softmax over HD_=1024 cols, in place on a bf16 buffer (f32 math).
__global__ __launch_bounds__(256)
void softmax_rows_bf16(uint16_t* __restrict__ wm) {
  uint16_t* p = wm + (long)blockIdx.x * HD_;
  __shared__ float red[256];
  int tid = threadIdx.x;
  float mx = -3.0e38f;
  for (int i = tid; i < HD_; i += 256) mx = fmaxf(mx, bf2f(p[i]));
  red[tid] = mx; __syncthreads();
  for (int s = 128; s > 0; s >>= 1) {
    if (tid < s) red[tid] = fmaxf(red[tid], red[tid + s]);
    __syncthreads();
  }
  mx = red[0]; __syncthreads();
  float sum = 0.0f;
  for (int i = tid; i < HD_; i += 256) sum += __expf(bf2f(p[i]) - mx);
  red[tid] = sum; __syncthreads();
  for (int s = 128; s > 0; s >>= 1) {
    if (tid < s) red[tid] += red[tid + s];
    __syncthreads();
  }
  float inv = 1.0f / red[0];
  for (int i = tid; i < HD_; i += 256) p[i] = f2bf(__expf(bf2f(p[i]) - mx) * inv);
}

// r_last[b,k] = dot(Hs[b,255,:], alpha[b,k,:])   (tiny: 67 MFLOP total)
__global__ __launch_bounds__(256)
void r_last_kernel(const uint16_t* __restrict__ HsB,
                   const uint16_t* __restrict__ alpha,
                   uint16_t* __restrict__ rbf) {
  int b = blockIdx.x, k = threadIdx.x;
  const uint16_t* hs = HsB + (long)b * (L_ * HD_) + 255 * HD_;
  const uint16_t* al = alpha + ((long)b * L_ + k) * HD_;
  float s = 0.0f;
  for (int h = 0; h < HD_; ++h) s += bf2f(hs[h]) * bf2f(al[h]);
  rbf[b * L_ + k] = f2bf(s);
}

__global__ void hstar_kernel(const float* __restrict__ WR,
                             const float* __restrict__ WHN,
                             float* __restrict__ hstar) {
  int idx = blockIdx.x * blockDim.x + threadIdx.x;
  if (idx < B_ * HD_) hstar[idx] = tanhf(WR[idx] + WHN[idx]);
}

__global__ __launch_bounds__(128)
void logits_kernel(const float* __restrict__ hstar,
                   const float* __restrict__ Ws_w,
                   const float* __restrict__ Ws_b,
                   float* __restrict__ y) {
  int b = threadIdx.x;
  const float* h = hstar + b * HD_;
  float lg[3];
#pragma unroll
  for (int cc = 0; cc < 3; ++cc) {
    float s = Ws_b[cc];
    const float* w = Ws_w + cc * HD_;
    for (int i = 0; i < HD_; ++i) s += h[i] * w[i];
    lg[cc] = s;
  }
  float mx = fmaxf(lg[0], fmaxf(lg[1], lg[2]));
  float e0 = __expf(lg[0] - mx), e1 = __expf(lg[1] - mx), e2 = __expf(lg[2] - mx);
  float inv = 1.0f / (e0 + e1 + e2);
  y[b * 3 + 0] = e0 * inv; y[b * 3 + 1] = e1 * inv; y[b * 3 + 2] = e2 * inv;
}

// ------------------------- conversion helpers ------------------------------
__global__ void cvt_bf16(uint16_t* __restrict__ dst, const float* __restrict__ src, long n) {
  long i = (long)blockIdx.x * blockDim.x + threadIdx.x;
  if (i < n) dst[i] = f2bf(src[i]);
}
__global__ void build_wcat(uint16_t* __restrict__ dst,
                           const float* __restrict__ wih,
                           const float* __restrict__ whh) {
  long i = (long)blockIdx.x * blockDim.x + threadIdx.x;
  if (i >= (long)G4_ * 2048) return;
  int n = (int)(i >> 11), k = (int)(i & 2047);
  float v = (k < 1024) ? wih[(long)n * 1024 + k] : whh[(long)n * 1024 + (k - 1024)];
  dst[i] = f2bf(v);
}
__global__ void bias_cat_kernel(float* __restrict__ dst,
                                const float* __restrict__ bih,
                                const float* __restrict__ bhh) {
  int i = blockIdx.x * blockDim.x + threadIdx.x;
  if (i < G4_) dst[i] = bih[i] + bhh[i];
}
__global__ void aspect_kernel(uint16_t* __restrict__ dst, const float* __restrict__ x) {
  int i = blockIdx.x * blockDim.x + threadIdx.x;   // 128*512
  if (i >= B_ * E_) return;
  int b = i >> 9, j = i & 511;
  dst[i] = f2bf(x[(long)b * (L_ * D_) + E_ + j]);  // x[b, 0, 512+j]
}

// ---------------------------------------------------------------------------
static inline long ceil_div(long a, long b) { return (a + b - 1) / b; }

extern "C" void kernel_launch(void* const* d_in, const int* in_sizes, int n_in,
                              void* d_out, int out_size, void* d_ws, size_t ws_size,
                              hipStream_t stream) {
  const float* x     = (const float*)d_in[0];
  const float* h0    = (const float*)d_in[1];
  const float* c0    = (const float*)d_in[2];
  const float* W_ih  = (const float*)d_in[3];
  const float* W_hh  = (const float*)d_in[4];
  const float* b_ih  = (const float*)d_in[5];
  const float* b_hh  = (const float*)d_in[6];
  const float* Whv_w = (const float*)d_in[7];
  const float* Whv_b = (const float*)d_in[8];
  const float* Wm_w  = (const float*)d_in[9];
  const float* Wm_b  = (const float*)d_in[10];
  const float* Wr_w  = (const float*)d_in[11];
  const float* Wr_b  = (const float*)d_in[12];
  const float* Whn_w = (const float*)d_in[13];
  const float* Whn_b = (const float*)d_in[14];
  const float* Ws_w  = (const float*)d_in[15];
  const float* Ws_b  = (const float*)d_in[16];

  float* y_out  = (float*)d_out;          // (128, 3)
  float* Hs_out = y_out + B_ * 3;         // (128, 256, 1024) f32

  // Deterministic workspace carve-up (256B aligned), ~334 MB total.
  char* w = (char*)d_ws;
  auto alloc = [&](size_t bytes) -> char* {
    char* p = w; w += (bytes + 255) & ~(size_t)255; return p;
  };
  uint16_t* wcat   = (uint16_t*)alloc((size_t)G4_ * 2048 * 2);   // [W_ih | W_hh] bf16
  float*    biascat= (float*)   alloc((size_t)G4_ * 4);
  uint16_t* x_bf   = (uint16_t*)alloc((size_t)B_ * L_ * D_ * 2);
  uint16_t* Hs_bf  = (uint16_t*)alloc((size_t)B_ * L_ * HD_ * 2);
  uint16_t* h_bf   = (uint16_t*)alloc((size_t)B_ * HD_ * 2);
  float*    c_st   = (float*)   alloc((size_t)B_ * HD_ * 4);
  float*    g_buf  = (float*)   alloc((size_t)B_ * G4_ * 4);
  uint16_t* asp_bf = (uint16_t*)alloc((size_t)B_ * E_ * 2);
  uint16_t* Whv_bf = (uint16_t*)alloc((size_t)HA_ * HA_ * 2);
  uint16_t* Wm_bf  = (uint16_t*)alloc((size_t)HD_ * HA_ * 2);
  uint16_t* Whn_bf = (uint16_t*)alloc((size_t)HD_ * HD_ * 2);
  uint16_t* Wr_bf  = (uint16_t*)alloc((size_t)HD_ * L_ * 2);
  uint16_t* M_bf   = (uint16_t*)alloc((size_t)BL_ * HA_ * 2);
  uint16_t* WM_bf  = (uint16_t*)alloc((size_t)BL_ * HD_ * 2);   // becomes alpha in place
  uint16_t* rl_bf  = (uint16_t*)alloc((size_t)B_ * L_ * 2);
  float*    WHN_f  = (float*)   alloc((size_t)B_ * HD_ * 4);
  float*    WR_f   = (float*)   alloc((size_t)B_ * HD_ * 4);
  float*    hstar  = (float*)   alloc((size_t)B_ * HD_ * 4);

  // ---- one-time conversions (bf16 weight/activation staging) ----
  build_wcat<<<dim3(ceil_div((long)G4_ * 2048, 256)), 256, 0, stream>>>(wcat, W_ih, W_hh);
  bias_cat_kernel<<<dim3(G4_ / 256), 256, 0, stream>>>(biascat, b_ih, b_hh);
  cvt_bf16<<<dim3(ceil_div((long)B_ * L_ * D_, 256)), 256, 0, stream>>>(x_bf, x, (long)B_ * L_ * D_);
  cvt_bf16<<<dim3(ceil_div((long)HA_ * HA_, 256)), 256, 0, stream>>>(Whv_bf, Whv_w, (long)HA_ * HA_);
  cvt_bf16<<<dim3(ceil_div((long)HD_ * HA_, 256)), 256, 0, stream>>>(Wm_bf, Wm_w, (long)HD_ * HA_);
  cvt_bf16<<<dim3(ceil_div((long)HD_ * HD_, 256)), 256, 0, stream>>>(Whn_bf, Whn_w, (long)HD_ * HD_);
  cvt_bf16<<<dim3(ceil_div((long)HD_ * L_, 256)), 256, 0, stream>>>(Wr_bf, Wr_w, (long)HD_ * L_);
  cvt_bf16<<<dim3(ceil_div((long)B_ * HD_, 256)), 256, 0, stream>>>(h_bf, h0, (long)B_ * HD_);
  hipMemcpyAsync(c_st, c0, (size_t)B_ * HD_ * 4, hipMemcpyDeviceToDevice, stream);
  aspect_kernel<<<dim3((B_ * E_) / 256), 256, 0, stream>>>(asp_bf, x);

  // ---- LSTM scan: g = [x_t | h] @ [W_ih|W_hh]^T + (b_ih+b_hh), then gates ----
  // MT=1 (16x64 tiles): keeps 512 waves in flight per latency-critical step.
  for (int t = 0; t < L_; ++t) {
    gemm_wmma<1><<<dim3(B_ / 16, G4_ / 64), 32, 0, stream>>>(
        x_bf + (long)t * D_, (long)L_ * D_,       // A1 = x[:,t,:], row stride L*D
        h_bf, (long)HD_, 0, D_,                   // A2 = h, split at K=1024
        wcat, 2048, biascat, 2048,
        g_buf, (uint16_t*)nullptr, (long)G4_, 0);
    lstm_gates<<<dim3((B_ * HD_) / 256), 256, 0, stream>>>(
        g_buf, c_st, Hs_out, h_bf, Hs_bf, t);
  }

  // ---- attention: M = tanh([Hs | aspect] @ Whv^T + b)  (32768 x 1536) ----
  // MT=2 (32x64 tiles): 8 WMMAs per 6 fragment loads, 24576 waves.
  gemm_wmma<2><<<dim3(BL_ / 32, HA_ / 64), 32, 0, stream>>>(
      Hs_bf, (long)HD_,
      asp_bf, (long)E_, 8, HD_,                   // aspect row = bl >> 8 = b
      Whv_bf, (long)HA_, Whv_b, HA_,
      (float*)nullptr, M_bf, (long)HA_, 1);

  // ---- WM = M @ Wm^T + b  (32768 x 1024), bf16 out for in-place softmax ----
  gemm_wmma<2><<<dim3(BL_ / 32, HD_ / 64), 32, 0, stream>>>(
      M_bf, (long)HA_, M_bf, (long)HA_, 0, HA_,
      Wm_bf, (long)HA_, Wm_b, HA_,
      (float*)nullptr, WM_bf, (long)HD_, 0);

  softmax_rows_bf16<<<dim3(BL_), 256, 0, stream>>>(WM_bf);  // alpha in place

  // ---- tail (only l = L-1 contributes to y) ----
  r_last_kernel<<<dim3(B_), 256, 0, stream>>>(Hs_bf, WM_bf, rl_bf);
  gemm_wmma<1><<<dim3(B_ / 16, HD_ / 64), 32, 0, stream>>>(     // WHN = h_n @ Whn^T + b
      Hs_bf + (long)(L_ - 1) * HD_, (long)L_ * HD_,
      Hs_bf, (long)HD_, 0, HD_,
      Whn_bf, (long)HD_, Whn_b, HD_,
      WHN_f, (uint16_t*)nullptr, (long)HD_, 0);
  gemm_wmma<1><<<dim3(B_ / 16, HD_ / 64), 32, 0, stream>>>(     // WR = r_last @ Wr^T + b
      rl_bf, (long)L_, rl_bf, (long)L_, 0, L_,
      Wr_bf, (long)L_, Wr_b, L_,
      WR_f, (uint16_t*)nullptr, (long)HD_, 0);
  hstar_kernel<<<dim3((B_ * HD_) / 256), 256, 0, stream>>>(WR_f, WHN_f, hstar);
  logits_kernel<<<1, 128, 0, stream>>>(hstar, Ws_w, Ws_b, y_out);
}
